// MultiHeadAttention_7636451852747
// MI455X (gfx1250) — matrix operand order repaired and checked by reading, not compiled, not run on hardware
//
#include <hip/hip_runtime.h>
#include <hip/hip_bf16.h>

// ---------------------------------------------------------------------------
// Fused MHA for gfx1250 (MI455X): f16 WMMA GEMMs + flash attention + layernorm
// Round 2: flash-attention kv-block widened 32 -> 64 to amortize softmax VALU
// (butterfly reductions + acc rescale are per-block costs) over 2x the WMMAs.
// ---------------------------------------------------------------------------

typedef _Float16 h8   __attribute__((ext_vector_type(8)));
typedef _Float16 v16h __attribute__((ext_vector_type(16)));
typedef float    v8f  __attribute__((ext_vector_type(8)));

#define D_MODEL 512
#define HEADS   8
#define HD      64
#define SEQ     4096
#define BATCH   2
#define ROWS    (BATCH * SEQ)        // 8192
#define HEAD_ELEMS  (SEQ * HD)       // 262144
#define BATCH_ELEMS (SEQ * D_MODEL)  // 2097152

// A operand, 16x32 f16 (M x K). ISA layout: lane l -> row (l&15);
// halves [0..7] = K[(l>>4)*8 + 0..7], halves [8..15] = K[(l>>4)*8 + 16..23].
static __device__ __forceinline__ v16h load_a16x32(const _Float16* __restrict__ base,
                                                   int ld, int lane) {
  const int row = lane & 15;
  const int kb  = (lane >> 4) << 3;
  const _Float16* p = base + (size_t)row * ld + kb;
  h8 lo = *(const h8*)(p);
  h8 hi = *(const h8*)(p + 16);
  v16h r;
#pragma unroll
  for (int i = 0; i < 8; ++i) { r[i] = lo[i]; r[i + 8] = hi[i]; }
  return r;
}

// B operand, 32x16 f16 (K x N), read from K-contiguous ("transposed") storage:
// baseT[n*ld + k]. ISA layout: lane l -> col (l&15); halves [0..7] = K[(l>>4)*8 + 0..7], halves [8..15] = K[(l>>4)*8 + 16..23].
static __device__ __forceinline__ v16h load_b32x16T(const _Float16* __restrict__ baseT,
                                                    int ld, int lane) {
  const int col = lane & 15;
  const int kb  = (lane >> 4) << 3;
  const _Float16* p = baseT + (size_t)col * ld + kb;
  h8 lo = *(const h8*)(p);
  h8 hi = *(const h8*)(p + 16);
  v16h r;
#pragma unroll
  for (int i = 0; i < 8; ++i) { r[i] = lo[i]; r[i + 8] = hi[i]; }
  return r;
}

static __device__ __forceinline__ v8f wmma_f16(v16h a, v16h b, v8f c) {
  return __builtin_amdgcn_wmma_f32_16x16x32_f16(false, a, false, b, (short)0, c,
                                                false, false);
}

// ---------------------------------------------------------------------------
// Kernel 1: x f32 -> f16
__global__ __launch_bounds__(256) void cvt_x(const float* __restrict__ x,
                                             _Float16* __restrict__ xh, int n) {
  int i = blockIdx.x * 256 + threadIdx.x;
  if (i < n) xh[i] = (_Float16)x[i];
}

// Kernel 2: W (k-major) -> Wt f16 (n-major, K contiguous) for all three mats
__global__ __launch_bounds__(256) void cvt_wT(const float* __restrict__ Wq,
                                              const float* __restrict__ Wk,
                                              const float* __restrict__ Wv,
                                              _Float16* __restrict__ wT) {
  int i   = blockIdx.x * 256 + threadIdx.x;   // i = n*512 + k
  int mat = blockIdx.y;
  const float* W = (mat == 0) ? Wq : (mat == 1) ? Wk : Wv;
  int n = i >> 9, k = i & 511;
  wT[(size_t)mat * (D_MODEL * D_MODEL) + i] = (_Float16)W[(size_t)k * D_MODEL + n];
}

// ---------------------------------------------------------------------------
// Kernel 3: fused QKV projection.  Each wave computes a 16x64 tile of x@W.
// Q,K stored row-major f16 [8192][512] (== [b][h][4096][64] flat).
// V stored TRANSPOSED per head: Vt[b][h][d2][s2] so P*V B-operands are contiguous.
__global__ __launch_bounds__(128) void qkv_gemm(
    const _Float16* __restrict__ xh, const _Float16* __restrict__ wT,
    const float* __restrict__ bq, const float* __restrict__ bk,
    const float* __restrict__ bv,
    _Float16* __restrict__ Qh, _Float16* __restrict__ Kh,
    _Float16* __restrict__ Vt) {
  const int lane  = threadIdx.x & 31;
  const int wave  = threadIdx.x >> 5;
  const int mat   = blockIdx.z;
  const int nbase = blockIdx.x * 64;
  const int mbase = blockIdx.y * 64 + wave * 16;
  const _Float16* wTm = wT + (size_t)mat * (D_MODEL * D_MODEL);
  const float* bias = (mat == 0) ? bq : (mat == 1) ? bk : bv;

  const v8f zero = {};
  v8f acc[4];
#pragma unroll
  for (int t = 0; t < 4; ++t) acc[t] = zero;

  for (int kk = 0; kk < D_MODEL; kk += 32) {
    __builtin_prefetch(xh + (size_t)mbase * D_MODEL + kk + 64, 0, 1);
    v16h a = load_a16x32(xh + (size_t)mbase * D_MODEL + kk, D_MODEL, lane);
#pragma unroll
    for (int t = 0; t < 4; ++t) {
      v16h b = load_b32x16T(wTm + (size_t)(nbase + 16 * t) * D_MODEL + kk,
                            D_MODEL, lane);
      acc[t] = wmma_f16(a, b, acc[t]);
    }
  }

  const int hf  = lane >> 4;
  const int col = lane & 15;
#pragma unroll
  for (int t = 0; t < 4; ++t) {
    const int n = nbase + 16 * t + col;
    const float bval = bias[n];
#pragma unroll
    for (int i = 0; i < 8; ++i) {
      const int m = mbase + i + 8 * hf;          // global row in [0,8192)
      const float v = acc[t][i] + bval;
      if (mat == 0) {
        Qh[(size_t)m * D_MODEL + n] = (_Float16)v;
      } else if (mat == 1) {
        Kh[(size_t)m * D_MODEL + n] = (_Float16)v;
      } else {
        const int b  = m >> 12;
        const int f  = ((m & 4095) << 9) + n;    // flat within batch
        const int hh = f >> 18;
        const int s2 = (f >> 6) & 4095;
        const int d2 = f & 63;
        Vt[(size_t)b * BATCH_ELEMS + (size_t)hh * HEAD_ELEMS +
           (size_t)d2 * SEQ + s2] = (_Float16)v;
      }
    }
  }
}

// ---------------------------------------------------------------------------
// Kernel 4: flash attention. 1 wave = 16 query rows; WG = 4 waves = 64 rows.
// kv processed in blocks of 64 -> 8 score WMMAs + 8 PV WMMAs per block, with
// one softmax reduction pass (butterflies amortized over 4 score tiles).
__global__ __launch_bounds__(128) void flash_attn(
    const _Float16* __restrict__ Qh, const _Float16* __restrict__ Kh,
    const _Float16* __restrict__ Vt, float* __restrict__ attn) {
  __shared__ _Float16 pbuf[4][16 * 64];   // per-wave P bounce (C-layout -> A-layout)

  const int lane  = threadIdx.x & 31;
  const int wave  = threadIdx.x >> 5;
  const int bh    = blockIdx.y;                 // b*8 + h
  const int qbase = blockIdx.x * 64 + wave * 16;

  const _Float16* Qhead = Qh + (size_t)bh * HEAD_ELEMS;
  const _Float16* Khead = Kh + (size_t)bh * HEAD_ELEMS;
  const _Float16* Vhead = Vt + (size_t)bh * HEAD_ELEMS;

  // Q tile 16x64, split into two 16x32 A operands (register-resident for whole loop)
  v16h a0 = load_a16x32(Qhead + (size_t)qbase * HD, HD, lane);
  v16h a1 = load_a16x32(Qhead + (size_t)qbase * HD + 32, HD, lane);

  const v8f zero = {};
  float mrow[8], lrow[8];
  v8f acc[4];
#pragma unroll
  for (int i = 0; i < 8; ++i) { mrow[i] = -1e30f; lrow[i] = 0.0f; }
#pragma unroll
  for (int t = 0; t < 4; ++t) acc[t] = zero;

  const int hf  = lane >> 4;
  const int col = lane & 15;
  const int prow_base = 8 * hf;
  const int arow = lane & 15;
  const int akb  = (lane >> 4) << 3;

  for (int kv = 0; kv < SEQ; kv += 64) {
    __builtin_prefetch(Khead + (size_t)(kv + 64) * HD, 0, 1);

    // scores S = Q * K^T * (1/8): four 16x16 key tiles (keys kv+16t .. kv+16t+15)
    v8f s[4];
#pragma unroll
    for (int t = 0; t < 4; ++t) {
      v16h bk0 = load_b32x16T(Khead + (size_t)(kv + 16 * t) * HD, HD, lane);
      v16h bk1 = load_b32x16T(Khead + (size_t)(kv + 16 * t) * HD + 32, HD, lane);
      s[t] = wmma_f16(a0, bk0, zero);
      s[t] = wmma_f16(a1, bk1, s[t]);
    }

    // online softmax: one butterfly pass covers all 64 keys of this block
    float p[4][8], fi[8];
#pragma unroll
    for (int i = 0; i < 8; ++i) {
      const float t0 = s[0][i] * 0.125f;        // 1/sqrt(64)
      const float t1 = s[1][i] * 0.125f;
      const float t2 = s[2][i] * 0.125f;
      const float t3 = s[3][i] * 0.125f;
      float bm = fmaxf(fmaxf(t0, t1), fmaxf(t2, t3));
#pragma unroll
      for (int d = 1; d <= 8; d <<= 1) bm = fmaxf(bm, __shfl_xor(bm, d, 32));
      const float mn = fmaxf(mrow[i], bm);
      fi[i]   = __expf(mrow[i] - mn);
      p[0][i] = __expf(t0 - mn);
      p[1][i] = __expf(t1 - mn);
      p[2][i] = __expf(t2 - mn);
      p[3][i] = __expf(t3 - mn);
      float rs = (p[0][i] + p[1][i]) + (p[2][i] + p[3][i]);
#pragma unroll
      for (int d = 1; d <= 8; d <<= 1) rs += __shfl_xor(rs, d, 32);
      lrow[i] = lrow[i] * fi[i] + rs;
      mrow[i] = mn;
    }
#pragma unroll
    for (int t = 0; t < 4; ++t)
#pragma unroll
      for (int i = 0; i < 8; ++i) acc[t][i] *= fi[i];

    // P (C-layout f32) -> LDS -> two A-layout 16x32 f16 operands
    // (same-wave LDS traffic: hardware keeps DS ops in order, no barrier needed)
#pragma unroll
    for (int t = 0; t < 4; ++t)
#pragma unroll
      for (int i = 0; i < 8; ++i)
        pbuf[wave][(prow_base + i) * 64 + 16 * t + col] = (_Float16)p[t][i];

    v16h pa0, pa1;
    {
      const h8 l0 = *(const h8*)&pbuf[wave][arow * 64 + akb];
      const h8 h0 = *(const h8*)&pbuf[wave][arow * 64 + akb + 16];
      const h8 l1 = *(const h8*)&pbuf[wave][arow * 64 + 32 + akb];
      const h8 h1 = *(const h8*)&pbuf[wave][arow * 64 + 32 + akb + 16];
#pragma unroll
      for (int i = 0; i < 8; ++i) {
        pa0[i] = l0[i]; pa0[i + 8] = h0[i];
        pa1[i] = l1[i]; pa1[i + 8] = h1[i];
      }
    }

    // O += P * V over K=64, V read from transposed storage (contiguous B operands)
#pragma unroll
    for (int t = 0; t < 4; ++t) {
      v16h vt0 = load_b32x16T(Vhead + (size_t)(16 * t) * SEQ + kv, SEQ, lane);
      v16h vt1 = load_b32x16T(Vhead + (size_t)(16 * t) * SEQ + kv + 32, SEQ, lane);
      acc[t] = wmma_f16(pa0, vt0, acc[t]);
      acc[t] = wmma_f16(pa1, vt1, acc[t]);
    }
  }

  float* outhead = attn + (size_t)bh * HEAD_ELEMS;
#pragma unroll
  for (int t = 0; t < 4; ++t) {
    const int d2 = 16 * t + col;
#pragma unroll
    for (int i = 0; i < 8; ++i) {
      const int q = qbase + i + prow_base;
      outhead[(size_t)q * HD + d2] = acc[t][i] / lrow[i];
    }
  }
}

// ---------------------------------------------------------------------------
// Kernel 5: y = LayerNorm(x + attn) * gamma + beta, one block per row of 512
__global__ __launch_bounds__(256) void add_layernorm(
    const float* __restrict__ x, const float* __restrict__ attn,
    const float* __restrict__ gamma, const float* __restrict__ beta,
    float* __restrict__ out) {
  __shared__ float ssum[8], ssq[8];
  const int row = blockIdx.x;
  const int tid = threadIdx.x;
  const float* xr = x + (size_t)row * D_MODEL;
  const float* ar = attn + (size_t)row * D_MODEL;

  const float h0 = xr[tid] + ar[tid];
  const float h1 = xr[tid + 256] + ar[tid + 256];
  float s  = h0 + h1;
  float sq = h0 * h0 + h1 * h1;
#pragma unroll
  for (int d = 16; d >= 1; d >>= 1) {
    s  += __shfl_xor(s, d, 32);
    sq += __shfl_xor(sq, d, 32);
  }
  const int lane = tid & 31, wv = tid >> 5;
  if (lane == 0) { ssum[wv] = s; ssq[wv] = sq; }
  __syncthreads();
  if (wv == 0) {
    float s2 = (lane < 8) ? ssum[lane] : 0.0f;
    float q2 = (lane < 8) ? ssq[lane] : 0.0f;
#pragma unroll
    for (int d = 4; d >= 1; d >>= 1) {
      s2 += __shfl_xor(s2, d, 32);
      q2 += __shfl_xor(q2, d, 32);
    }
    if (lane == 0) { ssum[0] = s2; ssq[0] = q2; }
  }
  __syncthreads();
  const float mean = ssum[0] * (1.0f / D_MODEL);
  const float var  = ssq[0] * (1.0f / D_MODEL) - mean * mean;
  const float rstd = rsqrtf(var + 1e-5f);
  out[(size_t)row * D_MODEL + tid] =
      (h0 - mean) * rstd * gamma[tid] + beta[tid];
  out[(size_t)row * D_MODEL + tid + 256] =
      (h1 - mean) * rstd * gamma[tid + 256] + beta[tid + 256];
}

// ---------------------------------------------------------------------------
extern "C" void kernel_launch(void* const* d_in, const int* in_sizes, int n_in,
                              void* d_out, int out_size, void* d_ws, size_t ws_size,
                              hipStream_t stream) {
  const float* x     = (const float*)d_in[0];
  const float* Wq    = (const float*)d_in[1];
  const float* bq    = (const float*)d_in[2];
  const float* Wk    = (const float*)d_in[3];
  const float* bk    = (const float*)d_in[4];
  const float* Wv    = (const float*)d_in[5];
  const float* bv    = (const float*)d_in[6];
  const float* gamma = (const float*)d_in[7];
  const float* beta  = (const float*)d_in[8];
  float* out = (float*)d_out;

  // workspace carve-up (~52 MB)
  _Float16* xh = (_Float16*)d_ws;                        // 8192*512 f16
  _Float16* wT = xh + (size_t)ROWS * D_MODEL;            // 3*512*512 f16 (transposed)
  _Float16* Qh = wT + 3 * D_MODEL * D_MODEL;             // 8192*512 f16
  _Float16* Kh = Qh + (size_t)ROWS * D_MODEL;            // 8192*512 f16
  _Float16* Vt = Kh + (size_t)ROWS * D_MODEL;            // 8192*512 f16 (per-head T)
  float*  attn = (float*)(Vt + (size_t)ROWS * D_MODEL);  // 8192*512 f32

  cvt_x<<<(ROWS * D_MODEL) / 256, 256, 0, stream>>>(x, xh, ROWS * D_MODEL);
  cvt_wT<<<dim3((D_MODEL * D_MODEL) / 256, 3), 256, 0, stream>>>(Wq, Wk, Wv, wT);
  qkv_gemm<<<dim3(D_MODEL / 64, ROWS / 64, 3), 128, 0, stream>>>(
      xh, wT, bq, bk, bv, Qh, Kh, Vt);
  flash_attn<<<dim3(SEQ / 64, BATCH * HEADS), 128, 0, stream>>>(Qh, Kh, Vt, attn);
  add_layernorm<<<ROWS, 256, 0, stream>>>(x, attn, gamma, beta, out);
}
